// Tokenizer_38697655337251
// MI455X (gfx1250) — 8-bit matrix operand order repaired and checked by reading, compiled, not run on hardware
//
#include <hip/hip_runtime.h>

typedef __attribute__((ext_vector_type(8))) int v8i;

#define B_     64
#define C_     8
#define S_     16384
#define WS_    32
#define STEP_  16
#define NW_    1022
#define EMBED_ 256
#define NROWS_ 4096
#define PM_    363u

__device__ __forceinline__ unsigned powmod363(int e) {
  unsigned r = 1u, b = PM_;
  while (e) {
    if (e & 1) r = (r * b) & 4095u;
    b = (b * b) & 4095u;
    e >>= 1;
  }
  return r;
}

__device__ __forceinline__ unsigned pack4(int4 q) {
  unsigned t0 = __builtin_amdgcn_perm((unsigned)q.y, (unsigned)q.x, 0x0C0C0400u);
  unsigned t1 = __builtin_amdgcn_perm((unsigned)q.w, (unsigned)q.z, 0x04000C0Cu);
  return t0 | t1;
}

__global__ __launch_bounds__(128) void Tokenizer_38697655337251_kernel(
    const int* __restrict__ x,
    const float* __restrict__ emb,
    float* __restrict__ out) {

  __shared__ __align__(16) unsigned bpack[16][64];
  __shared__ unsigned tokbuf[4][32];

  const int tid  = threadIdx.x;
  const int lane = tid & 31;
  const int wave = tid >> 5;
  const int b    = blockIdx.y;

  {
    const int t64 = tid & 63;
    const int nn  = tid >> 6;
    unsigned p3 = powmod363(253 - 4 * t64);
    unsigned p2 = (p3 * PM_) & 4095u;
    unsigned p1 = (p2 * PM_) & 4095u;
    unsigned p0 = (p1 * PM_) & 4095u;
    unsigned b0, b1, b2, b3;
    if (nn == 0) { b0 = p0 & 0xFFu; b1 = p1 & 0xFFu; b2 = p2 & 0xFFu; b3 = p3 & 0xFFu; }
    else         { b0 = p0 >> 8;    b1 = p1 >> 8;    b2 = p2 >> 8;    b3 = p3 >> 8;    }
    bpack[nn][t64] = b0 | (b1 << 8) | (b2 << 16) | (b3 << 24);
#pragma unroll
    for (int k = 1; k < 8; ++k) {
      const int idx = tid + 128 * k;
      bpack[idx >> 6][idx & 63] = 0u;
    }
  }
  __syncthreads();

  const int tile = blockIdx.x * 4 + wave;
  const int w0   = tile * 16;

  const int m   = lane & 15;
  const int n   = lane & 15;
  const int grp = lane >> 4;
  const int wl  = min(w0 + m, NW_ - 1);
  const int s0  = wl * STEP_;
  const int jb  = grp * 16;

  __builtin_prefetch(x + ((size_t)b * C_) * S_ + s0, 0, 0);

  v8i acc = {0, 0, 0, 0, 0, 0, 0, 0};

#pragma unroll
  for (int kk = 0; kk < 4; ++kk) {
    const int  c0 = 2 * kk;
    const int* r0 = x + ((size_t)b * C_ + c0) * S_ + s0;
    const int* r1 = r0 + S_;

    int4 q0 = *(const int4*)(r0 + jb);
    int4 q1 = *(const int4*)(r0 + jb + 4);
    int4 q2 = *(const int4*)(r0 + jb + 8);
    int4 q3 = *(const int4*)(r0 + jb + 12);
    int4 q4 = *(const int4*)(r1 + jb);
    int4 q5 = *(const int4*)(r1 + jb + 4);
    int4 q6 = *(const int4*)(r1 + jb + 8);
    int4 q7 = *(const int4*)(r1 + jb + 12);
    v8i a;
    a[0] = (int)pack4(q0); a[1] = (int)pack4(q1);
    a[2] = (int)pack4(q2); a[3] = (int)pack4(q3);
    a[4] = (int)pack4(q4); a[5] = (int)pack4(q5);
    a[6] = (int)pack4(q6); a[7] = (int)pack4(q7);

    const unsigned* bp = &bpack[n][kk * 16 + grp * 4];
    uint4 blo = *(const uint4*)bp;
    uint4 bhi = *(const uint4*)(bp + 8);
    v8i bf;
    bf[0] = (int)blo.x; bf[1] = (int)blo.y; bf[2] = (int)blo.z; bf[3] = (int)blo.w;
    bf[4] = (int)bhi.x; bf[5] = (int)bhi.y; bf[6] = (int)bhi.z; bf[7] = (int)bhi.w;

    acc = __builtin_amdgcn_wmma_i32_16x16x64_iu8(false, a, false, bf, acc,
                                                 false, false);
  }

  if (n < 2) {
    const int base = n * 16 + grp * 8;
#pragma unroll
    for (int i = 0; i < 8; ++i) tokbuf[wave][base + i] = (unsigned)acc[i];
  }
  __syncthreads();

  const float4* embv = (const float4*)emb;
  float4*       outv = (float4*)out;
#pragma unroll 1
  for (int mm = 0; mm < 16; ++mm) {
    const int w = w0 + mm;
    if (w >= NW_) break;
    const unsigned tok =
        (tokbuf[wave][mm] + (tokbuf[wave][16 + mm] << 8)) & 4095u;
    const float4* src = embv + (size_t)tok * (EMBED_ / 4);
    float4*       dst = outv + ((size_t)b * NW_ + w) * (EMBED_ / 4);
    dst[lane]      = src[lane];
    dst[lane + 32] = src[lane + 32];
  }
}

extern "C" void kernel_launch(void* const* d_in, const int* in_sizes, int n_in,
                              void* d_out, int out_size, void* d_ws, size_t ws_size,
                              hipStream_t stream) {
  const int*   x   = (const int*)d_in[0];
  const float* emb = (const float*)d_in[1];
  float*       out = (float*)d_out;
  (void)in_sizes; (void)n_in; (void)out_size; (void)d_ws; (void)ws_size;

  dim3 grid(16, B_, 1);
  dim3 block(128, 1, 1);
  Tokenizer_38697655337251_kernel<<<grid, block, 0, stream>>>(x, emb, out);
}
